// LocationWiseMoETransformerLayer_32676111188735
// MI455X (gfx1250) — matrix operand order repaired and checked by reading, not compiled, not run on hardware
//
#include <hip/hip_runtime.h>
#include <hip/hip_bf16.h>

// ---------------------------------------------------------------------------
// CDNA5 (gfx1250) wave32 WMMA types
// ---------------------------------------------------------------------------
typedef __attribute__((ext_vector_type(16))) __bf16       v16bf;
typedef __attribute__((ext_vector_type(8)))  float        v8f;
typedef __attribute__((ext_vector_type(4)))  unsigned int u32x4;

union AFrag { v16bf v; u32x4 q[2]; };           // 16 bf16 per lane (A or B operand)
union CFrag { v8f   v; float f[8]; };           // 16x16 f32 accumulator (8 VGPRs)

__device__ __forceinline__ unsigned short f2bf(float f) {
  unsigned int u = __float_as_uint(f);
  u += 0x7fffu + ((u >> 16) & 1u);              // round-to-nearest-even
  return (unsigned short)(u >> 16);
}

// ---------------------------------------------------------------------------
// Tensor Data Mover (async tensor DMA into LDS), guarded so both device
// toolchains and the host pass compile. Descriptor per CDNA5 ISA.
//   - 2D tile: tile_dim0 = 32 bf16 (16 DWORDs) along K, tile_dim1 = rows
//   - LDS pad: +8 DWORDs after every 16 DWORDs -> 96B row stride, which is
//     exactly the padded LDS layout the WMMA fragment b128 loads expect.
// ---------------------------------------------------------------------------
#if defined(__AMDGCN__) && __has_builtin(__builtin_amdgcn_tensor_load_to_lds) && \
    __has_builtin(__builtin_amdgcn_s_wait_tensorcnt)
#define USE_TDM 1
#endif

#ifdef USE_TDM
typedef unsigned int tdm_u4 __attribute__((ext_vector_type(4)));
typedef int          tdm_i8 __attribute__((ext_vector_type(8)));
typedef int          tdm_i4 __attribute__((ext_vector_type(4)));

__device__ __forceinline__ void tdm_load_tile(unsigned lds_off,
                                              const unsigned short* gptr,
                                              int rows, int Kstride) {
  unsigned long long ga = (unsigned long long)gptr;
  tdm_u4 g0;
  g0.x = 1u;                                                  // count=1 (valid)
  g0.y = lds_off;                                             // lds_addr
  g0.z = (unsigned)ga;                                        // global_addr lo
  g0.w = (unsigned)((ga >> 32) & 0x01ffffffull) | (2u << 30); // addr hi | type=2
  tdm_i8 g1;
  // wg_mask=0 | data_size=1(2B)<<16 | pad_en<<20 | pad_interval=3(16dw)<<22 |
  // pad_amount=7(8dw)<<25
  g1[0] = (1 << 16) | (1 << 20) | (3 << 22) | (7 << 25);
  g1[1] = (int)(((unsigned)Kstride & 0xffffu) << 16);         // tensor_dim0 lo16
  g1[2] = (int)(((unsigned)Kstride >> 16) | ((unsigned)rows << 16)); // t0 hi | t1 lo
  g1[3] = (int)(32u << 16);                                   // t1 hi=0 | tile_dim0=32
  g1[4] = rows;                                               // tile_dim1 | tile_dim2=0
  g1[5] = Kstride;                                            // dim0_stride lo32
  g1[6] = 0;                                                  // dim0_stride hi | dim1_stride lo
  g1[7] = 0;                                                  // dim1_stride hi
  tdm_i4 gz = {0, 0, 0, 0};
#if defined(__clang_major__) && (__clang_major__ >= 23)
  tdm_i8 gz8 = {0, 0, 0, 0, 0, 0, 0, 0};
  __builtin_amdgcn_tensor_load_to_lds(g0, g1, gz, gz, gz8, 0);
#else
  __builtin_amdgcn_tensor_load_to_lds(g0, g1, gz, gz, 0);
#endif
}
#endif  // USE_TDM

// ---------------------------------------------------------------------------
// fp32 -> bf16 bulk conversion
// ---------------------------------------------------------------------------
__global__ void __launch_bounds__(256)
f32_to_bf16_kernel(const float* __restrict__ in, unsigned short* __restrict__ out, long n) {
  long i = (long)blockIdx.x * blockDim.x + threadIdx.x;
  long stride = (long)gridDim.x * blockDim.x;
  for (; i < n; i += stride) out[i] = f2bf(in[i]);
}

// ---------------------------------------------------------------------------
// Batched tiled transpose + fp32->bf16: in [z][R][C] f32 -> out [z][C][R] bf16.
// Coalesced read and write via 32x33 LDS tile.
// ---------------------------------------------------------------------------
__global__ void __launch_bounds__(256)
transpose_f32_to_bf16_kernel(const float* __restrict__ in, unsigned short* __restrict__ out,
                             int R, int C) {
  __shared__ float tile[32][33];
  long zoff = (long)blockIdx.z * R * C;
  int c0 = blockIdx.x * 32, r0 = blockIdx.y * 32;
  int tx = threadIdx.x & 31, ty = threadIdx.x >> 5;  // 32x8
#pragma unroll
  for (int i = 0; i < 4; ++i)
    tile[ty + i * 8][tx] = in[zoff + (long)(r0 + ty + i * 8) * C + c0 + tx];
  __syncthreads();
#pragma unroll
  for (int i = 0; i < 4; ++i)
    out[zoff + (long)(c0 + ty + i * 8) * R + r0 + tx] = f2bf(tile[tx][ty + i * 8]);
}

// ---------------------------------------------------------------------------
// LayerNorm over E=512, bf16 output. transposed=1 writes location-major [L,B,E].
// ---------------------------------------------------------------------------
__global__ void __launch_bounds__(256)
layernorm_bf16_kernel(const float* __restrict__ x, const float* __restrict__ g,
                      const float* __restrict__ bta, unsigned short* __restrict__ out,
                      int L, int transposed) {
  const int E = 512;
  int token = blockIdx.x;
  int tid = threadIdx.x;
  const float* row = x + (long)token * E;
  float v0 = row[tid];
  float v1 = row[tid + 256];
  __shared__ float red[256];
  red[tid] = v0 + v1;
  __syncthreads();
  for (int o = 128; o > 0; o >>= 1) { if (tid < o) red[tid] += red[tid + o]; __syncthreads(); }
  float mean = red[0] * (1.0f / E);
  __syncthreads();
  float d0 = v0 - mean, d1 = v1 - mean;
  red[tid] = d0 * d0 + d1 * d1;
  __syncthreads();
  for (int o = 128; o > 0; o >>= 1) { if (tid < o) red[tid] += red[tid + o]; __syncthreads(); }
  float rstd = rsqrtf(red[0] * (1.0f / E) + 1e-5f);
  float y0 = d0 * rstd * g[tid] + bta[tid];
  float y1 = d1 * rstd * g[tid + 256] + bta[tid + 256];
  long base;
  if (transposed) {
    int b = token / L, l = token % L;
    int Brows = gridDim.x / L;
    base = ((long)l * Brows + b) * E;
  } else {
    base = (long)token * E;
  }
  out[base + tid]       = f2bf(y0);
  out[base + tid + 256] = f2bf(y1);
}

// ---------------------------------------------------------------------------
// bf16 WMMA GEMM:  C[z] = act(A[z] @ B[e]^T + bias[e]) (+res)
//   A: [M,K] row-major bf16.  B: [N,K] row-major bf16 (weights pre-transposed).
//   Block tile 64(M) x 128(N), 8 waves each computing 32x32, K chunks of 32.
//   Tiles staged into LDS by the Tensor Data Mover (TENSORcnt) when available.
//   RESMODE 0: plain (OUTBF selects bf16/f32 out)
//   RESMODE 1: +res (f32, same layout), f32 out
//   RESMODE 2: +res at ((row*resL + z)*N + col), f32 out at same index
// ---------------------------------------------------------------------------
template <int GELU_, int OUTBF, int RESMODE>
__global__ void __launch_bounds__(256)
gemm_wmma_kernel(const unsigned short* __restrict__ A, const unsigned short* __restrict__ Bm,
                 const float* __restrict__ bias, const float* __restrict__ res,
                 void* __restrict__ Cout, int M, int N, int K,
                 long aBatch, long bBatch, long biasBatch, long cBatch,
                 const int* __restrict__ assign, int resL) {
  __shared__ unsigned short as[64 * 48];   // [row][k] k-contiguous, 96B row stride
  __shared__ unsigned short bs[128 * 48];  // [col][k] k-contiguous, 96B row stride
  const int z = blockIdx.z;
  const int expert = assign ? assign[z] : z;
  const unsigned short* Ab = A + (long)z * aBatch;
  const unsigned short* Bb = Bm + (long)expert * bBatch;
  const float* biasb = bias + (long)expert * biasBatch;

  const int tid = threadIdx.x;
  const int wave = tid >> 5, lane = tid & 31;
  const int laneLo = lane & 15, hi = lane >> 4;
  const int wm = (wave >> 2) * 32, wn = (wave & 3) * 32;
  const int blockM = blockIdx.y * 64, blockN = blockIdx.x * 128;

  CFrag acc[2][2];
#pragma unroll
  for (int mt = 0; mt < 2; ++mt)
#pragma unroll
    for (int nt = 0; nt < 2; ++nt)
#pragma unroll
      for (int r = 0; r < 8; ++r) acc[mt][nt].f[r] = 0.f;

#ifdef USE_TDM
  const unsigned as_off = (unsigned)(uintptr_t)&as[0];
  const unsigned bs_off = (unsigned)(uintptr_t)&bs[0];
#endif

  for (int kc = 0; kc < K; kc += 32) {
    __syncthreads();  // previous iteration's fragment reads done
#ifdef USE_TDM
    // Tensor Data Mover: DMA both tiles into LDS with the padded layout.
    if (wave == 0) {
      tdm_load_tile(as_off, &Ab[(long)blockM * K + kc], 64, K);
    } else if (wave == 1) {
      tdm_load_tile(bs_off, &Bb[(long)blockN * K + kc], 128, K);
    }
    if (wave < 2) __builtin_amdgcn_s_wait_tensorcnt(0);
#else
    {  // A tile: 64x32, 8 bf16 (one b128) per thread
      int idx = tid * 8;
      int row = idx >> 5, c0 = idx & 31;
      const unsigned short* src = &Ab[(long)(blockM + row) * K + kc + c0];
      *(u32x4*)&as[row * 48 + c0] = *(const u32x4*)src;
      if (kc + 32 < K) __builtin_prefetch(src + 32, 0, 0);
    }
#pragma unroll
    for (int i = 0; i < 2; ++i) {  // B tile: 128x32
      int idx = (tid + i * 256) * 8;
      int n = idx >> 5, k0 = idx & 31;
      const unsigned short* src = &Bb[(long)(blockN + n) * K + kc + k0];
      *(u32x4*)&bs[n * 48 + k0] = *(const u32x4*)src;
      if (kc + 32 < K) __builtin_prefetch(src + 32, 0, 0);
    }
#endif
    __syncthreads();

    AFrag af[2], bf2[2];
#pragma unroll
    for (int mt = 0; mt < 2; ++mt) {   // A frag: K = (e<8?e:e+8) + 8*hi
      int row = wm + mt * 16 + laneLo;
      af[mt].q[0] = *(const u32x4*)&as[row * 48 + hi * 8];
      af[mt].q[1] = *(const u32x4*)&as[row * 48 + hi * 8 + 16];
    }
#pragma unroll
    for (int nt = 0; nt < 2; ++nt) {   // B frag: K = (e<8?e:e+8) + 8*hi
      int col = wn + nt * 16 + laneLo;
      bf2[nt].q[0] = *(const u32x4*)&bs[col * 48 + hi * 8];
      bf2[nt].q[1] = *(const u32x4*)&bs[col * 48 + hi * 8 + 16];
    }
#pragma unroll
    for (int mt = 0; mt < 2; ++mt)
#pragma unroll
      for (int nt = 0; nt < 2; ++nt)
        acc[mt][nt].v = __builtin_amdgcn_wmma_f32_16x16x32_bf16(
            false, af[mt].v, false, bf2[nt].v, (short)0, acc[mt][nt].v, false, false);
  }

  // Epilogue (C layout: row = r + 8*hi, col = laneLo within each 16x16 tile)
#pragma unroll
  for (int mt = 0; mt < 2; ++mt)
#pragma unroll
    for (int nt = 0; nt < 2; ++nt)
#pragma unroll
      for (int r = 0; r < 8; ++r) {
        int row = blockM + wm + mt * 16 + r + hi * 8;
        int col = blockN + wn + nt * 16 + laneLo;
        if (row < M && col < N) {
          float v = acc[mt][nt].f[r] + biasb[col];
          if (GELU_) v = 0.5f * v * (1.0f + erff(v * 0.70710678118f));
          if (RESMODE == 1) {
            long idx = (long)z * cBatch + (long)row * N + col;
            ((float*)Cout)[idx] = v + res[idx];
          } else if (RESMODE == 2) {
            long idx = ((long)row * resL + z) * (long)N + col;
            ((float*)Cout)[idx] = v + res[idx];
          } else {
            long idx = (long)z * cBatch + (long)row * N + col;
            if (OUTBF) ((unsigned short*)Cout)[idx] = f2bf(v);
            else       ((float*)Cout)[idx] = v;
          }
        }
      }
}

// ---------------------------------------------------------------------------
// Fused attention: one (b,h) per 128-thread block (4 waves).
// Scores (64x64) via WMMA, in-register softmax (16-lane shfl reductions match
// the C-matrix lane map), P@V via WMMA with V staged transposed in LDS.
// ---------------------------------------------------------------------------
__global__ void __launch_bounds__(128)
attention_kernel(const unsigned short* __restrict__ qkv, unsigned short* __restrict__ o) {
  const int H = 8;
  const int b = blockIdx.x / H;
  const int h = blockIdx.x % H;
  __shared__ unsigned short qs[64 * 72];
  __shared__ unsigned short ks[64 * 72];
  __shared__ unsigned short vsT[64 * 72];  // [d][k]
  __shared__ unsigned short ps[64 * 72];
  const int tid = threadIdx.x;
  for (int i = tid; i < 64 * 64; i += 128) {
    int row = i >> 6, d = i & 63;
    long base = ((long)(b * 64 + row)) * 1536 + h * 64 + d;
    qs[row * 72 + d]  = qkv[base];
    ks[row * 72 + d]  = qkv[base + 512];
    vsT[d * 72 + row] = qkv[base + 1024];
  }
  __syncthreads();

  const int wave = tid >> 5, lane = tid & 31;
  const int laneLo = lane & 15, hi = lane >> 4;
  const int m0 = wave * 16;

  CFrag sacc[4];
#pragma unroll
  for (int t = 0; t < 4; ++t)
#pragma unroll
    for (int r = 0; r < 8; ++r) sacc[t].f[r] = 0.f;

#pragma unroll
  for (int kc = 0; kc < 64; kc += 32) {  // S = Q @ K^T  (K dim = Dh)
    AFrag a;
    a.q[0] = *(const u32x4*)&qs[(m0 + laneLo) * 72 + kc + hi * 8];
    a.q[1] = *(const u32x4*)&qs[(m0 + laneLo) * 72 + kc + hi * 8 + 16];
#pragma unroll
    for (int t = 0; t < 4; ++t) {
      AFrag bb;
      int col = t * 16 + laneLo;  // key index
      bb.q[0] = *(const u32x4*)&ks[col * 72 + kc + hi * 8];
      bb.q[1] = *(const u32x4*)&ks[col * 72 + kc + hi * 8 + 16];
      sacc[t].v = __builtin_amdgcn_wmma_f32_16x16x32_bf16(
          false, a.v, false, bb.v, (short)0, sacc[t].v, false, false);
    }
  }

  // row softmax: each 16-lane group of VGPR r holds one full row across 4 tiles
#pragma unroll
  for (int r = 0; r < 8; ++r) {
    float mx = -3.4e38f;
#pragma unroll
    for (int t = 0; t < 4; ++t) { sacc[t].f[r] *= 0.125f; mx = fmaxf(mx, sacc[t].f[r]); }
    for (int msk = 8; msk; msk >>= 1) mx = fmaxf(mx, __shfl_xor(mx, msk, 32));
    float sm = 0.f;
#pragma unroll
    for (int t = 0; t < 4; ++t) { float e = __expf(sacc[t].f[r] - mx); sacc[t].f[r] = e; sm += e; }
    for (int msk = 8; msk; msk >>= 1) sm += __shfl_xor(sm, msk, 32);
    float inv = 1.0f / sm;
    int prow = m0 + r + hi * 8;
#pragma unroll
    for (int t = 0; t < 4; ++t)
      ps[prow * 72 + t * 16 + laneLo] = f2bf(sacc[t].f[r] * inv);
  }
  __syncthreads();

  CFrag oacc[4];
#pragma unroll
  for (int t = 0; t < 4; ++t)
#pragma unroll
    for (int r = 0; r < 8; ++r) oacc[t].f[r] = 0.f;

#pragma unroll
  for (int kc = 0; kc < 64; kc += 32) {  // O = P @ V  (K dim = key index)
    AFrag a;
    a.q[0] = *(const u32x4*)&ps[(m0 + laneLo) * 72 + kc + hi * 8];
    a.q[1] = *(const u32x4*)&ps[(m0 + laneLo) * 72 + kc + hi * 8 + 16];
#pragma unroll
    for (int t = 0; t < 4; ++t) {
      AFrag bb;
      int col = t * 16 + laneLo;  // d
      bb.q[0] = *(const u32x4*)&vsT[col * 72 + kc + hi * 8];
      bb.q[1] = *(const u32x4*)&vsT[col * 72 + kc + hi * 8 + 16];
      oacc[t].v = __builtin_amdgcn_wmma_f32_16x16x32_bf16(
          false, a.v, false, bb.v, (short)0, oacc[t].v, false, false);
    }
  }
#pragma unroll
  for (int t = 0; t < 4; ++t)
#pragma unroll
    for (int r = 0; r < 8; ++r) {
      int row = m0 + r + hi * 8;
      int d = t * 16 + laneLo;
      o[((long)(b * 64 + row)) * 512 + h * 64 + d] = f2bf(oacc[t].f[r]);
    }
}

// ---------------------------------------------------------------------------
// Orchestration
// ---------------------------------------------------------------------------
extern "C" void kernel_launch(void* const* d_in, const int* in_sizes, int n_in,
                              void* d_out, int out_size, void* d_ws, size_t ws_size,
                              hipStream_t stream) {
  const float* src        = (const float*)d_in[0];
  const float* in_proj_w  = (const float*)d_in[1];
  const float* in_proj_b  = (const float*)d_in[2];
  const float* out_proj_w = (const float*)d_in[3];
  const float* out_proj_b = (const float*)d_in[4];
  const float* ln1_g      = (const float*)d_in[5];
  const float* ln1_b      = (const float*)d_in[6];
  const float* ln2_g      = (const float*)d_in[7];
  const float* ln2_b      = (const float*)d_in[8];
  const float* W1         = (const float*)d_in[9];
  const float* b1         = (const float*)d_in[10];
  const float* W2         = (const float*)d_in[11];
  const float* b2         = (const float*)d_in[12];
  const int*   assign     = (const int*)d_in[13];
  (void)in_sizes; (void)n_in; (void)out_size; (void)ws_size;

  const int Bsz = 512, L = 64, E = 512, F = 2048, H = 8;
  const int T = Bsz * L;  // 32768 tokens

  char* ws = (char*)d_ws;
  size_t off = 0;
  auto alloc = [&](size_t bytes) {
    void* p = ws + off;
    off = (off + bytes + 255) & ~(size_t)255;
    return p;
  };
  unsigned short* w_in_bf  = (unsigned short*)alloc((size_t)3 * E * E * 2);  // [3E,E]
  unsigned short* w_out_bf = (unsigned short*)alloc((size_t)E * E * 2);      // [E,E]
  unsigned short* W1T_bf   = (unsigned short*)alloc((size_t)L * E * F * 2);  // [L,F,E]
  unsigned short* W2T_bf   = (unsigned short*)alloc((size_t)L * F * E * 2);  // [L,E,F]
  unsigned short* x1_bf    = (unsigned short*)alloc((size_t)T * E * 2);
  unsigned short* qkv_bf   = (unsigned short*)alloc((size_t)T * 3 * E * 2);
  unsigned short* o_bf     = (unsigned short*)alloc((size_t)T * E * 2);
  float*          x_f32    = (float*)alloc((size_t)T * E * 4);
  unsigned short* x2_bf    = (unsigned short*)alloc((size_t)T * E * 2);      // [L,B,E]
  unsigned short* h_bf     = (unsigned short*)alloc((size_t)T * F * 2);      // [L,B,F]

  // 1) stage weights as bf16; W1/W2 transposed so all GEMM B operands are [N,K]
  f32_to_bf16_kernel<<<2048, 256, 0, stream>>>(in_proj_w, w_in_bf, (long)3 * E * E);
  f32_to_bf16_kernel<<<2048, 256, 0, stream>>>(out_proj_w, w_out_bf, (long)E * E);
  transpose_f32_to_bf16_kernel<<<dim3(F / 32, E / 32, L), 256, 0, stream>>>(W1, W1T_bf, E, F);
  transpose_f32_to_bf16_kernel<<<dim3(E / 32, F / 32, L), 256, 0, stream>>>(W2, W2T_bf, F, E);

  // 2) LN1 -> bf16 token-major
  layernorm_bf16_kernel<<<T, 256, 0, stream>>>(src, ln1_g, ln1_b, x1_bf, L, 0);

  // 3) QKV = x1 @ in_proj_w^T + b   [T,1536] bf16
  gemm_wmma_kernel<0, 1, 0><<<dim3(3 * E / 128, T / 64, 1), 256, 0, stream>>>(
      x1_bf, w_in_bf, in_proj_b, nullptr, qkv_bf, T, 3 * E, E, 0, 0, 0, 0, nullptr, 0);

  // 4) fused attention -> o_bf [T,E]
  attention_kernel<<<Bsz * H, 128, 0, stream>>>(qkv_bf, o_bf);

  // 5) x = o @ out_proj_w^T + b + src   [T,E] f32
  gemm_wmma_kernel<0, 0, 1><<<dim3(E / 128, T / 64, 1), 256, 0, stream>>>(
      o_bf, w_out_bf, out_proj_b, src, x_f32, T, E, E, 0, 0, 0, (long)T * E, nullptr, 0);

  // 6) LN2 -> bf16 location-major [L,B,E]
  layernorm_bf16_kernel<<<T, 256, 0, stream>>>(x_f32, ln2_g, ln2_b, x2_bf, L, 1);

  // 7) h = gelu(x2 @ W1[assign[l]] + b1[assign[l]])  batched over l, bf16 [L,B,F]
  gemm_wmma_kernel<1, 1, 0><<<dim3(F / 128, Bsz / 64, L), 256, 0, stream>>>(
      x2_bf, W1T_bf, b1, nullptr, h_bf, Bsz, F, E,
      (long)Bsz * E, (long)E * F, (long)F, (long)Bsz * F, assign, 0);

  // 8) out = h @ W2[assign[l]] + b2[assign[l]] + x   (transposed store -> [B,L,E])
  gemm_wmma_kernel<0, 0, 2><<<dim3(E / 128, Bsz / 64, L), 256, 0, stream>>>(
      h_bf, W2T_bf, b2, x_f32, d_out, Bsz, E, F,
      (long)Bsz * F, (long)F * E, (long)E, 0, assign, L);
}
